// MulScoreDecoder_86809878987246
// MI455X (gfx1250) — matrix operand order repaired and checked by reading, not compiled, not run on hardware
//
#include <hip/hip_runtime.h>
#include <hip/hip_bf16.h>

// ---------------------------------------------------------------------------
// Problem constants (from reference)
// ---------------------------------------------------------------------------
#define N_S   10
#define BB    32
#define CC    768
#define HH    24
#define WW    24
#define HT    8
#define WT    8
#define HEADS 12
#define DH    64            // CC / HEADS
#define POOL  4
#define NB    (N_S * BB)    // 320 tokens per "row"
#define T0    (POOL * POOL) // 16 pooled tokens
#define T1    (HT * WT)     // 64 template tokens
#define NTOK0 (NB * T0)     // 5120
#define NTOK1 (NB * T1)     // 20480
#define WXY_STRIDE 200

typedef _Float16 v16h __attribute__((ext_vector_type(16)));
typedef _Float16 v8h  __attribute__((ext_vector_type(8)));
typedef float    v8f  __attribute__((ext_vector_type(8)));
typedef unsigned int u32x4 __attribute__((ext_vector_type(4)));
typedef int          i32x4 __attribute__((ext_vector_type(4)));
typedef int          i32x8 __attribute__((ext_vector_type(8)));

#if defined(__has_builtin)
#  if __has_builtin(__builtin_amdgcn_tensor_load_to_lds) && \
      __has_builtin(__builtin_amdgcn_s_wait_tensorcnt)
#    define USE_TDM 1
#  endif
#endif
#ifndef USE_TDM
#  define USE_TDM 0
#endif

// ---------------------------------------------------------------------------
// TDM: 1-D contiguous panel load (nelem 2-byte elements) into LDS.
// Descriptor packing per CDNA5 D# layout (group0: count/lds/global/type,
// group1: data_size=2B, tensor_dim0=tile_dim0=nelem, tile_dim1=1).
// ---------------------------------------------------------------------------
#if USE_TDM
__device__ __forceinline__ void tdm_load_panel(const void* gaddr, unsigned lds_off,
                                               int nelem) {
    unsigned long long ga = (unsigned long long)(size_t)gaddr;
    u32x4 g0;
    g0[0] = 1u;                                              // count=1, user mode
    g0[1] = lds_off;                                         // LDS byte address
    g0[2] = (unsigned)ga;                                    // global_addr[31:0]
    g0[3] = (unsigned)((ga >> 32) & 0x01FFFFFFu) | 0x80000000u; // [56:32] | type=2
    i32x8 g1;
    g1[0] = 1 << 16;                                         // data_size = 2 bytes
    g1[1] = (nelem & 0xFFFF) << 16;                          // tensor_dim0[15:0]
    g1[2] = ((nelem >> 16) & 0xFFFF) | (1 << 16);            // dim0 hi | tensor_dim1=1
    g1[3] = (nelem & 0xFFFF) << 16;                          // tile_dim0
    g1[4] = 1;                                               // tile_dim1=1
    g1[5] = nelem;                                           // tensor_dim0_stride
    g1[6] = 0;
    g1[7] = 0;
    i32x4 g2 = {0, 0, 0, 0};
    i32x4 g3 = {0, 0, 0, 0};
#if __clang_major__ >= 23
    i32x8 g4 = {0, 0, 0, 0, 0, 0, 0, 0};
    __builtin_amdgcn_tensor_load_to_lds(g0, g1, g2, g3, g4, 0);
#else
    __builtin_amdgcn_tensor_load_to_lds(g0, g1, g2, g3, 0);
#endif
}
#endif

// ---------------------------------------------------------------------------
// Block reduction helper (wave32 aware)
// ---------------------------------------------------------------------------
__device__ __forceinline__ float blockReduceSum256(float v) {
    for (int off = 16; off; off >>= 1) v += __shfl_xor(v, off, 32);
    __shared__ float s[8];
    __shared__ float tot;
    int lane = threadIdx.x & 31, wid = threadIdx.x >> 5;
    if (lane == 0) s[wid] = v;
    __syncthreads();
    if (threadIdx.x < 8) {
        float w = s[threadIdx.x];
        for (int off = 4; off; off >>= 1) w += __shfl_xor(w, off, 32);
        if (threadIdx.x == 0) tot = w;
    }
    __syncthreads();
    float r = tot;
    __syncthreads();
    return r;
}

// ---------------------------------------------------------------------------
// 1) Per-ROI PrROI weight tables: wy[4][24], wx[4][24], 1/area
// ---------------------------------------------------------------------------
__device__ __forceinline__ float Gfun(float t) {
    float tc = fminf(1.f, fmaxf(-1.f, t));
    return 0.5f + tc * (1.f - 0.5f * fabsf(tc));
}

__global__ void roi_weights_kernel(const float* __restrict__ boxes, float* __restrict__ wxy) {
    int r = blockIdx.x * blockDim.x + threadIdx.x;
    if (r >= NB) return;
    float x0 = boxes[r * 4 + 0] * (float)WW;
    float y0 = boxes[r * 4 + 1] * (float)WW;
    float x1 = boxes[r * 4 + 2] * (float)WW;
    float y1 = boxes[r * 4 + 3] * (float)WW;
    float bw = (x1 - x0) * 0.25f;
    float bh = (y1 - y0) * 0.25f;
    float* wy = wxy + (size_t)r * WXY_STRIDE;
    float* wx = wy + 96;
    for (int p = 0; p < POOL; ++p) {
        float ey0 = y0 + p * bh, ey1 = ey0 + bh;
        float ex0 = x0 + p * bw, ex1 = ex0 + bw;
        for (int i = 0; i < HH; ++i) {
            float fi = (float)i;
            wy[p * HH + i] = Gfun(ey1 - fi) - Gfun(ey0 - fi);
            wx[p * HH + i] = Gfun(ex1 - fi) - Gfun(ex0 - fi);
        }
    }
    wxy[(size_t)r * WXY_STRIDE + 192] = 1.f / fmaxf(bw * bh, 1e-8f);
}

// ---------------------------------------------------------------------------
// 2) PrROI pooling: one wave per (roi, channel); feats read exactly once.
// ---------------------------------------------------------------------------
__global__ __launch_bounds__(256) void prroi_pool_kernel(
    const float* __restrict__ feats, const float* __restrict__ wxy,
    _Float16* __restrict__ sbf16)
{
    __shared__ float sw[WXY_STRIDE];
    int r = blockIdx.x;
    for (int i = threadIdx.x; i < 193; i += 256) sw[i] = wxy[(size_t)r * WXY_STRIDE + i];
    __syncthreads();
    int wid = threadIdx.x >> 5, lane = threadIdx.x & 31;
    int c = blockIdx.y * 8 + wid;
    const float* f = feats + ((size_t)r * CC + c) * (HH * WW);

    float colsum[POOL] = {0.f, 0.f, 0.f, 0.f};
    for (int h = 0; h < HH; ++h) {
        float v = (lane < WW) ? f[h * WW + lane] : 0.f;
        colsum[0] += sw[0 * HH + h] * v;
        colsum[1] += sw[1 * HH + h] * v;
        colsum[2] += sw[2 * HH + h] * v;
        colsum[3] += sw[3 * HH + h] * v;
    }
    float acc[16];
#pragma unroll
    for (int p = 0; p < POOL; ++p)
#pragma unroll
        for (int q = 0; q < POOL; ++q) {
            float wxv = (lane < WW) ? sw[96 + q * HH + lane] : 0.f;
            acc[p * 4 + q] = colsum[p] * wxv;
        }
#pragma unroll
    for (int j = 0; j < 16; ++j)
        for (int off = 16; off; off >>= 1) acc[j] += __shfl_xor(acc[j], off, 32);
    if (lane == 0) {
        float inv = sw[192];
#pragma unroll
        for (int j = 0; j < 16; ++j)
            sbf16[((size_t)r * T0 + j) * CC + c] = (_Float16)(acc[j] * inv);
    }
}

// ---------------------------------------------------------------------------
// 3) Template (N,b,C,64) -> token-major f16 via LDS transpose
// ---------------------------------------------------------------------------
__global__ __launch_bounds__(256) void tmpl_pack_kernel(
    const float* __restrict__ in, _Float16* __restrict__ out)
{
    __shared__ float tile[64 * 65];
    int nb = blockIdx.x;
    const float* src = in + (size_t)nb * CC * T1;
    _Float16* dst = out + (size_t)nb * T1 * CC;
    for (int c0 = 0; c0 < CC; c0 += 64) {
        __syncthreads();
        for (int i = threadIdx.x; i < 64 * 64; i += 256) {
            int cc = i >> 6, t = i & 63;
            tile[cc * 65 + t] = src[(size_t)(c0 + cc) * T1 + t];
        }
        __syncthreads();
        for (int i = threadIdx.x; i < 64 * 64; i += 256) {
            int t = i >> 6, cc = i & 63;
            dst[(size_t)t * CC + c0 + cc] = (_Float16)tile[cc * 65 + t];
        }
    }
}

// ---------------------------------------------------------------------------
// 4) Pack f32 weight (K x N) into WMMA B-fragment layout:
//    Bp[((kt*nTiles+nt)*32 + lane)*16 + j], k = kt*32+(j>>3)*16+(lane>>4)*8+(j&7), n = nt*16+(lane&15)
// ---------------------------------------------------------------------------
__global__ void pack_b_kernel(const float* __restrict__ W, _Float16* __restrict__ Bp,
                              int K, int N)
{
    int idx = blockIdx.x * blockDim.x + threadIdx.x;
    if (idx >= K * N) return;
    int j    = idx & 15;
    int lane = (idx >> 4) & 31;
    int tile = idx >> 9;
    int nTiles = N >> 4;
    int kt = tile / nTiles, nt = tile - kt * nTiles;
    int k = kt * 32 + (j >> 3) * 16 + (lane >> 4) * 8 + (j & 7);
    int n = nt * 16 + (lane & 15);
    Bp[idx] = (_Float16)W[(size_t)k * N + n];
}

// ---------------------------------------------------------------------------
// 5) WMMA GEMM, K known at compile time (KDIM=768 everywhere here).
//    Block tile 64x64, 8 waves (4M x 2N), 2 WMMAs per wave per k-step.
//    Double-buffered LDS; A staged cooperatively with prefetch, B panel
//    (4KB contiguous in packed layout) moved by the TDM engine
//    (tensor_load_to_lds + s_wait_tensorcnt) overlapping compute.
//    Last k-step peeled; unroll-2 makes buffer parity compile-time.
// ---------------------------------------------------------------------------
#define TILE_M 64
#define TILE_N 64
#define LDA_P  40   // padded A row stride (halves)

template <int KDIM>
__global__ __launch_bounds__(256) void gemm_wmma_kernel(
    const _Float16* __restrict__ A, const _Float16* __restrict__ Bp,
    const float* __restrict__ bias,
    float* __restrict__ Cf32, _Float16* __restrict__ Cf16,
    int M, int N, int relu)
{
    constexpr int NK = KDIM >> 5;
    __shared__ alignas(32) _Float16 ldsA[2][TILE_M * LDA_P];
    __shared__ alignas(32) _Float16 ldsB[2][4 * 512];   // 4 n-tiles * 512 halves
    int tid  = threadIdx.x;
    int lane = tid & 31;
    int wid  = tid >> 5;
    int mw   = wid & 3;       // 0..3 -> M sub-tile
    int nw   = wid >> 2;      // 0..1 -> N sub-tile pair
    int mBlock = blockIdx.x * TILE_M;
    int nTiles = N >> 4;
    int ntBase = blockIdx.y * 4;     // first 16-col tile of this block

    int rowL = tid >> 2;             // cooperative A-tile load: 64 rows x 32 halves
    int colL = (tid & 3) * 8;
    const _Float16* aRow = A + (size_t)(mBlock + rowL) * KDIM + colL;
    int aRowFrag = mw * 16 + (lane & 15);
    int kb = (lane >> 4) * 8;
    int aoff = aRowFrag * LDA_P + kb;
    int boff0 = (nw * 2 + 0) * 512 + lane * 16;
    int boff1 = (nw * 2 + 1) * 512 + lane * 16;

    // ---- prologue: stage k-step 0 into buffer 0
    {
        v8h a0 = *(const v8h*)(aRow);
        *(v8h*)&ldsA[0][rowL * LDA_P + colL] = a0;
        const _Float16* panel = Bp + ((size_t)ntBase) * 512;
#if USE_TDM
        if (wid == 0) {
            tdm_load_panel(panel, (unsigned)(size_t)&ldsB[0][0], 4 * 512);
            __builtin_amdgcn_s_wait_tensorcnt(0);
        }
#else
        *(v8h*)&ldsB[0][tid * 8] = *(const v8h*)(panel + tid * 8);
#endif
        __syncthreads();
    }

    v8f acc0 = {}, acc1 = {};
    // ---- steady state: always stages k-step kt+1 while computing kt
#pragma unroll 2
    for (int kt = 0; kt < NK - 1; ++kt) {
        const int cur = kt & 1, nxt = cur ^ 1;
        v8h aNext = *(const v8h*)(aRow + (kt + 1) * 32);
        const _Float16* panel = Bp + ((size_t)(kt + 1) * nTiles + ntBase) * 512;
#if USE_TDM
        if (wid == 0)
            tdm_load_panel(panel, (unsigned)(size_t)&ldsB[nxt][0], 4 * 512);
#else
        *(v8h*)&ldsB[nxt][tid * 8] = *(const v8h*)(panel + tid * 8);
#endif
        if (kt + 2 < NK) __builtin_prefetch(aRow + (kt + 2) * 32, 0, 1);

        union { v16h v; v8h h[2]; } af;
        af.h[0] = *(const v8h*)&ldsA[cur][aoff];        // K kb..kb+7
        af.h[1] = *(const v8h*)&ldsA[cur][aoff + 16];   // K 16+kb..
        v16h b0 = *(const v16h*)&ldsB[cur][boff0];
        v16h b1 = *(const v16h*)&ldsB[cur][boff1];

        acc0 = __builtin_amdgcn_wmma_f32_16x16x32_f16(
            false, af.v, false, b0, (short)0, acc0, false, false);
        acc1 = __builtin_amdgcn_wmma_f32_16x16x32_f16(
            false, af.v, false, b1, (short)0, acc1, false, false);

        *(v8h*)&ldsA[nxt][rowL * LDA_P + colL] = aNext;
#if USE_TDM
        if (wid == 0) __builtin_amdgcn_s_wait_tensorcnt(0);
#endif
        __syncthreads();
    }
    // ---- peeled final k-step (buffer parity is compile-time)
    {
        constexpr int cur = (NK - 1) & 1;
        union { v16h v; v8h h[2]; } af;
        af.h[0] = *(const v8h*)&ldsA[cur][aoff];
        af.h[1] = *(const v8h*)&ldsA[cur][aoff + 16];
        v16h b0 = *(const v16h*)&ldsB[cur][boff0];
        v16h b1 = *(const v16h*)&ldsB[cur][boff1];
        acc0 = __builtin_amdgcn_wmma_f32_16x16x32_f16(
            false, af.v, false, b0, (short)0, acc0, false, false);
        acc1 = __builtin_amdgcn_wmma_f32_16x16x32_f16(
            false, af.v, false, b1, (short)0, acc1, false, false);
    }

    // C layout: VGPR r -> M = r (lanes 0-15) / r+8 (lanes 16-31), N = lane&15
    int row0 = mBlock + mw * 16 + (lane >> 4) * 8;
    int col  = ntBase * 16 + nw * 32 + (lane & 15);
    float bv0 = bias ? bias[col] : 0.f;
    float bv1 = bias ? bias[col + 16] : 0.f;
#pragma unroll
    for (int r = 0; r < 8; ++r) {
        float v0 = acc0[r] + bv0;
        float v1 = acc1[r] + bv1;
        if (relu) { v0 = fmaxf(v0, 0.f); v1 = fmaxf(v1, 0.f); }
        size_t i0 = (size_t)(row0 + r) * N + col;
        if (Cf32) { Cf32[i0] = v0; Cf32[i0 + 16] = v1; }
        if (Cf16) { Cf16[i0] = (_Float16)v0; Cf16[i0 + 16] = (_Float16)v1; }
    }
}

// ---------------------------------------------------------------------------
// 6) LayerNorm over C=768 per token
// ---------------------------------------------------------------------------
__global__ __launch_bounds__(256) void layernorm_kernel(
    const float* __restrict__ x, const float* __restrict__ g, const float* __restrict__ b,
    float* __restrict__ of32, _Float16* __restrict__ of16)
{
    int t = blockIdx.x;
    const float* xi = x + (size_t)t * CC;
    float s = 0.f;
    for (int i = threadIdx.x; i < CC; i += 256) s += xi[i];
    float mean = blockReduceSum256(s) * (1.f / CC);
    float v = 0.f;
    for (int i = threadIdx.x; i < CC; i += 256) { float d = xi[i] - mean; v += d * d; }
    float var = blockReduceSum256(v) * (1.f / CC);
    float inv = rsqrtf(var + 1e-5f);
    for (int i = threadIdx.x; i < CC; i += 256) {
        float o = (xi[i] - mean) * inv * g[i] + b[i];
        if (of32) of32[(size_t)t * CC + i] = o;
        if (of16) of16[(size_t)t * CC + i] = (_Float16)o;
    }
}

// ---------------------------------------------------------------------------
// 7) Small matvec: out[n] = bias[n] + sum_k x[k]*W[k*N+n]  (q0 projection)
// ---------------------------------------------------------------------------
__global__ void matvec_kernel(const float* __restrict__ x, const float* __restrict__ W,
                              const float* __restrict__ bias, float* __restrict__ out,
                              int K, int N)
{
    int n = blockIdx.x * blockDim.x + threadIdx.x;
    if (n >= N) return;
    float acc = bias[n];
    for (int k = 0; k < K; ++k) acc += x[k] * W[(size_t)k * N + n];
    out[n] = acc;
}

// ---------------------------------------------------------------------------
// 8) Per-(token, head) attention over T in {16, 64}; block = 64 threads
// ---------------------------------------------------------------------------
__global__ __launch_bounds__(64) void attention_kernel(
    const float* __restrict__ Q, int qStride,
    const _Float16* __restrict__ Kt, const _Float16* __restrict__ Vt,
    _Float16* __restrict__ O, int T, float scale)
{
    __shared__ float qsh[DH];
    __shared__ float sattn[64];
    __shared__ float mx, ssum;
    int token = blockIdx.x;
    int head  = blockIdx.y;
    int d     = threadIdx.x;
    qsh[d] = Q[(size_t)token * qStride + head * DH + d];
    __syncthreads();
    float s = -1e30f;
    if (d < T) {
        const _Float16* k = Kt + ((size_t)token * T + d) * CC + head * DH;
        float a = 0.f;
        for (int i = 0; i < DH; ++i) a += qsh[i] * (float)k[i];
        s = a * scale;
    }
    sattn[d] = s;
    __syncthreads();
    if (d == 0) {
        float m = -1e30f;
        for (int t = 0; t < T; ++t) m = fmaxf(m, sattn[t]);
        mx = m;
    }
    __syncthreads();
    float e = (d < T) ? __expf(sattn[d] - mx) : 0.f;
    sattn[d] = e;
    __syncthreads();
    if (d == 0) {
        float s2 = 0.f;
        for (int t = 0; t < T; ++t) s2 += sattn[t];
        ssum = s2;
    }
    __syncthreads();
    float inv = 1.f / ssum;
    float o = 0.f;
    const _Float16* v = Vt + (size_t)token * T * CC + head * DH + d;
    for (int t = 0; t < T; ++t) o += sattn[t] * (float)v[(size_t)t * CC];
    O[(size_t)token * CC + head * DH + d] = (_Float16)(o * inv);
}

// ---------------------------------------------------------------------------
// 9) N-mixing: newx[n][b][c] = bmix[n] + sum_m ln[m][b][c]*Wmix[m][n]
// ---------------------------------------------------------------------------
__global__ void mix_kernel(const float* __restrict__ ln, const float* __restrict__ Wmix,
                           const float* __restrict__ bmix,
                           float* __restrict__ xo, _Float16* __restrict__ xo16)
{
    int idx = blockIdx.x * blockDim.x + threadIdx.x;
    if (idx >= N_S * BB * CC) return;
    int c = idx % CC;
    int b = (idx / CC) % BB;
    int n = idx / (CC * BB);
    float acc = bmix[n];
#pragma unroll
    for (int m = 0; m < N_S; ++m)
        acc += ln[((size_t)m * BB + b) * CC + c] * Wmix[m * N_S + n];
    size_t o = ((size_t)n * BB + b) * CC + c;
    xo[o] = acc;
    xo16[o] = (_Float16)acc;
}

// ---------------------------------------------------------------------------
// 10) Head: out[token] = b3 + sum_c H[token][c]*W3[c]
// ---------------------------------------------------------------------------
__global__ __launch_bounds__(256) void head_kernel(
    const _Float16* __restrict__ H, const float* __restrict__ W3,
    const float* __restrict__ b3, float* __restrict__ out)
{
    int token = blockIdx.x;
    float s = 0.f;
    for (int i = threadIdx.x; i < CC; i += 256)
        s += (float)H[(size_t)token * CC + i] * W3[i];
    float tot = blockReduceSum256(s);
    if (threadIdx.x == 0) out[token] = tot + b3[0];
}

// ---------------------------------------------------------------------------
// Launch
// ---------------------------------------------------------------------------
static inline void* bump(char*& p, size_t bytes) {
    void* r = (void*)p;
    p += (bytes + 255) & ~(size_t)255;
    return r;
}

extern "C" void kernel_launch(void* const* d_in, const int* in_sizes, int n_in,
                              void* d_out, int out_size, void* d_ws, size_t ws_size,
                              hipStream_t stream) {
    const float* search_feats  = (const float*)d_in[0];
    const float* template_feat = (const float*)d_in[1];
    const float* search_boxes  = (const float*)d_in[2];
    const float* score_token   = (const float*)d_in[3];
    const float* ln1_g = (const float*)d_in[4];
    const float* ln1_b = (const float*)d_in[5];
    const float* Wq = (const float*)d_in[6];
    const float* bq = (const float*)d_in[7];
    const float* Wk = (const float*)d_in[8];
    const float* bk = (const float*)d_in[9];
    const float* Wv = (const float*)d_in[10];
    const float* bv = (const float*)d_in[11];
    const float* Wo = (const float*)d_in[12];
    const float* bo = (const float*)d_in[13];
    const float* ln2_g = (const float*)d_in[14];
    const float* ln2_b = (const float*)d_in[15];
    const float* Wmix = (const float*)d_in[16];
    const float* bmix = (const float*)d_in[17];
    const float* W1 = (const float*)d_in[18];
    const float* b1 = (const float*)d_in[19];
    const float* W2 = (const float*)d_in[20];
    const float* b2 = (const float*)d_in[21];
    const float* W3 = (const float*)d_in[22];
    const float* b3 = (const float*)d_in[23];

    char* p = (char*)d_ws;
    float*     wxy    = (float*)    bump(p, (size_t)NB * WXY_STRIDE * 4);
    _Float16*  sbf16  = (_Float16*) bump(p, (size_t)NTOK0 * CC * 2);
    _Float16*  tmpl16 = (_Float16*) bump(p, (size_t)NTOK1 * CC * 2);
    size_t wsz = (size_t)CC * CC * 2;
    _Float16* pWk0 = (_Float16*)bump(p, wsz);
    _Float16* pWv0 = (_Float16*)bump(p, wsz);
    _Float16* pWk1 = (_Float16*)bump(p, wsz);
    _Float16* pWv1 = (_Float16*)bump(p, wsz);
    _Float16* pWq1 = (_Float16*)bump(p, wsz);
    _Float16* pWo0 = (_Float16*)bump(p, wsz);
    _Float16* pWo1 = (_Float16*)bump(p, wsz);
    _Float16* pW1  = (_Float16*)bump(p, wsz);
    _Float16* pW2  = (_Float16*)bump(p, wsz);
    _Float16* K0 = (_Float16*)bump(p, (size_t)NTOK0 * CC * 2);
    _Float16* V0 = (_Float16*)bump(p, (size_t)NTOK0 * CC * 2);
    _Float16* K1 = (_Float16*)bump(p, (size_t)NTOK1 * CC * 2);
    _Float16* V1 = (_Float16*)bump(p, (size_t)NTOK1 * CC * 2);
    float*    x0    = (float*)   bump(p, (size_t)CC * 4);
    float*    q0    = (float*)   bump(p, (size_t)CC * 4);
    _Float16* oatt  = (_Float16*)bump(p, (size_t)NB * CC * 2);
    float*    Oraw  = (float*)   bump(p, (size_t)NB * CC * 4);
    float*    lnbuf = (float*)   bump(p, (size_t)NB * CC * 4);
    float*    xf    = (float*)   bump(p, (size_t)NB * CC * 4);
    _Float16* xh    = (_Float16*)bump(p, (size_t)NB * CC * 2);
    float*    Q1    = (float*)   bump(p, (size_t)NB * CC * 4);
    _Float16* h1    = (_Float16*)bump(p, (size_t)NB * CC * 2);
    _Float16* h2    = (_Float16*)bump(p, (size_t)NB * CC * 2);

    const float scale = 0.03608439182435161f;  // 768^-0.5

    // Stage 1: pooling + template tokenization
    roi_weights_kernel<<<(NB + 63) / 64, 64, 0, stream>>>(search_boxes, wxy);
    prroi_pool_kernel<<<dim3(NB, CC / 8), 256, 0, stream>>>(search_feats, wxy, sbf16);
    tmpl_pack_kernel<<<NB, 256, 0, stream>>>(template_feat, tmpl16);

    // Stage 2: pack weights into WMMA B-fragment layout
    int pgrid = (CC * CC + 255) / 256;
    pack_b_kernel<<<pgrid, 256, 0, stream>>>(Wk,           pWk0, CC, CC);
    pack_b_kernel<<<pgrid, 256, 0, stream>>>(Wv,           pWv0, CC, CC);
    pack_b_kernel<<<pgrid, 256, 0, stream>>>(Wk + CC * CC, pWk1, CC, CC);
    pack_b_kernel<<<pgrid, 256, 0, stream>>>(Wv + CC * CC, pWv1, CC, CC);
    pack_b_kernel<<<pgrid, 256, 0, stream>>>(Wq + CC * CC, pWq1, CC, CC);
    pack_b_kernel<<<pgrid, 256, 0, stream>>>(Wo,           pWo0, CC, CC);
    pack_b_kernel<<<pgrid, 256, 0, stream>>>(Wo + CC * CC, pWo1, CC, CC);
    pack_b_kernel<<<pgrid, 256, 0, stream>>>(W1,           pW1,  CC, CC);
    pack_b_kernel<<<pgrid, 256, 0, stream>>>(W2,           pW2,  CC, CC);

    // Stage 3: K/V projections (WMMA, TDM-fed B panels)
    gemm_wmma_kernel<CC><<<dim3(NTOK0 / TILE_M, CC / TILE_N), 256, 0, stream>>>(
        sbf16, pWk0, bk, nullptr, K0, NTOK0, CC, 0);
    gemm_wmma_kernel<CC><<<dim3(NTOK0 / TILE_M, CC / TILE_N), 256, 0, stream>>>(
        sbf16, pWv0, bv, nullptr, V0, NTOK0, CC, 0);
    gemm_wmma_kernel<CC><<<dim3(NTOK1 / TILE_M, CC / TILE_N), 256, 0, stream>>>(
        tmpl16, pWk1, bk + CC, nullptr, K1, NTOK1, CC, 0);
    gemm_wmma_kernel<CC><<<dim3(NTOK1 / TILE_M, CC / TILE_N), 256, 0, stream>>>(
        tmpl16, pWv1, bv + CC, nullptr, V1, NTOK1, CC, 0);

    // Stage 4: score token -> ln1 -> q0 (single 768-vector, shared by all)
    layernorm_kernel<<<1, 256, 0, stream>>>(score_token, ln1_g, ln1_b, x0, nullptr);
    matvec_kernel<<<(CC + 255) / 256, 256, 0, stream>>>(x0, Wq, bq, q0, CC, CC);

    // Branch 0: pooled-search cross attention
    attention_kernel<<<dim3(NB, HEADS), 64, 0, stream>>>(q0, 0, K0, V0, oatt, T0, scale);
    gemm_wmma_kernel<CC><<<dim3(NB / TILE_M, CC / TILE_N), 256, 0, stream>>>(
        oatt, pWo0, bo, Oraw, nullptr, NB, CC, 0);
    layernorm_kernel<<<NB, 256, 0, stream>>>(Oraw, ln2_g, ln2_b, lnbuf, nullptr);
    mix_kernel<<<(N_S * BB * CC + 255) / 256, 256, 0, stream>>>(lnbuf, Wmix, bmix, xf, xh);

    // Branch 1: template cross attention
    gemm_wmma_kernel<CC><<<dim3(NB / TILE_M, CC / TILE_N), 256, 0, stream>>>(
        xh, pWq1, bq + CC, Q1, nullptr, NB, CC, 0);
    attention_kernel<<<dim3(NB, HEADS), 64, 0, stream>>>(Q1, CC, K1, V1, oatt, T1, scale);
    gemm_wmma_kernel<CC><<<dim3(NB / TILE_M, CC / TILE_N), 256, 0, stream>>>(
        oatt, pWo1, bo + CC, Oraw, nullptr, NB, CC, 0);
    layernorm_kernel<<<NB, 256, 0, stream>>>(Oraw, ln2_g + CC, ln2_b + CC, lnbuf, nullptr);
    mix_kernel<<<(N_S * BB * CC + 255) / 256, 256, 0, stream>>>(
        lnbuf, Wmix + N_S * N_S, bmix + N_S, xf, xh);

    // MLP head
    gemm_wmma_kernel<CC><<<dim3(NB / TILE_M, CC / TILE_N), 256, 0, stream>>>(
        xh, pW1, b1, nullptr, h1, NB, CC, 1);
    gemm_wmma_kernel<CC><<<dim3(NB / TILE_M, CC / TILE_N), 256, 0, stream>>>(
        h1, pW2, b2, nullptr, h2, NB, CC, 1);
    head_kernel<<<NB, 256, 0, stream>>>(h2, W3, b3, (float*)d_out);

    (void)in_sizes; (void)n_in; (void)out_size; (void)ws_size;
}
